// MaskedAttention_24653112279246
// MI455X (gfx1250) — matrix operand order repaired and checked by reading, not compiled, not run on hardware
//
#include <hip/hip_runtime.h>
#include <hip/hip_bf16.h>
#include <math.h>

// ---------------------------------------------------------------------------
// MaskedAttention forward for MI455X (gfx1250), wave32 + WMMA f16 16x16x32.
// B=2, T=2048, D=1024, H=16, DK=64.  SCALE = 1/sqrt(64) = 0.125.
// ---------------------------------------------------------------------------

typedef _Float16 half_t;
typedef __attribute__((ext_vector_type(16))) _Float16 v16h;
typedef __attribute__((ext_vector_type(8)))  _Float16 v8h;
typedef __attribute__((ext_vector_type(8)))  float    v8f;
typedef int v4i_raw __attribute__((vector_size(16)));   // matches builtin param

#define CB 2
#define CT 2048
#define CD 1024
#define CH 16
#define CDK 64
#define CM (CB*CT)          // 4096 rows
#define QK_SCALE 0.125f

#define LQ 72               // f16 LDS tile stride (halfs): 16B-aligned, low-conflict
#define LS 65               // f32 LDS tile stride (floats): conflict-free columns
#define LG 40               // f16 GEMM stage stride (halfs)

// ---- gfx1250 async global->LDS path (ASYNCcnt), with safe fallback --------
#if defined(__gfx1250__) && __has_builtin(__builtin_amdgcn_global_load_async_to_lds_b128)
#define ASYNC_LDS 1
#endif

static __device__ __forceinline__ void async_b128(const void* g, void* l) {
#ifdef ASYNC_LDS
  // as(1) ptr is 64-bit; as(3) ptr is 32-bit and flat-LDS low 32 bits are the
  // LDS offset (ISA 10.2 aperture mapping), so integer casts are value-correct.
  __builtin_amdgcn_global_load_async_to_lds_b128(
      (__attribute__((address_space(1))) v4i_raw*)(uintptr_t)g,
      (__attribute__((address_space(3))) v4i_raw*)(unsigned int)(uintptr_t)l,
      0, 0);
#else
  *(v8h*)l = *(const v8h*)g;
#endif
}

#ifdef ASYNC_LDS
#define ASYNC_WAIT() asm volatile("s_wait_asynccnt 0x0" ::: "memory")
#else
#define ASYNC_WAIT()
#endif

static __device__ __forceinline__ v8f wmma_f16(v16h a, v16h b, v8f c) {
  // D = A(16x32 f16) * B(32x16 f16) + C(16x16 f32)
  return __builtin_amdgcn_wmma_f32_16x16x32_f16(
      false, a, false, b, (short)0, c, false, false);
}

// A fragment: 16(M) x 32(K) tile, row-major f16, stride lda (halfs).
// ISA 7.12.2: lanes 0-15 -> M=lane, halfs {K 0..7, K 16..23};
//             lanes 16-31 -> M=lane-16, halfs {K 8..15, K 24..31}.
static __device__ __forceinline__ v16h load_frag_a(const half_t* A, int lda, int lane) {
  int m  = lane & 15;
  int k0 = (lane >> 4) << 3;
  const v8h lo = *(const v8h*)(A + m * lda + k0);
  const v8h hi = *(const v8h*)(A + m * lda + k0 + 16);
  v16h r;
#pragma unroll
  for (int i = 0; i < 8; ++i) { r[i] = lo[i]; r[i + 8] = hi[i]; }
  return r;
}

// B fragment: operand B(32K x 16N) supplied as row-major [N][K] tile, stride
// ldb. Lane holds column n=lane&15, K {8h..8h+7, 16+8h..16+8h+7}, h=lane>>4.
static __device__ __forceinline__ v16h load_frag_b(const half_t* Bt, int ldb, int lane) {
  int n  = lane & 15;
  int k0 = (lane >> 4) << 3;
  const v8h lo = *(const v8h*)(Bt + n * ldb + k0);
  const v8h hi = *(const v8h*)(Bt + n * ldb + k0 + 16);
  v16h r;
#pragma unroll
  for (int i = 0; i < 8; ++i) { r[i] = lo[i]; r[i + 8] = hi[i]; }
  return r;
}

// ---------------------------------------------------------------------------
__global__ void cvt_f32_to_f16_kernel(const float* __restrict__ src,
                                      half_t* __restrict__ dst, int n) {
  for (int i = blockIdx.x * blockDim.x + threadIdx.x; i < n;
       i += gridDim.x * blockDim.x)
    dst[i] = (half_t)src[i];
}

// ---------------------------------------------------------------------------
// Y = A(f16 [4096 x 1024]) @ W^T + bias.  64x64 tile, 8 waves, double-buffered
// LDS staging (1 barrier / K-step).
// mode 0/1: per-head L2 norm, f16 out [b][h][t][dk]    (Q / K)
// mode 2:   f16 out transposed [b][h][dk][t]           (V^T)
// mode 3:   fp32 out = 0.5f * Y to d_out [B,T,D]       (final projection)
// ---------------------------------------------------------------------------
__global__ __launch_bounds__(256) void gemm64_kernel(
    const half_t* __restrict__ A, const half_t* __restrict__ Wf,
    const float* __restrict__ bias, half_t* __restrict__ outH,
    float* __restrict__ outF, int mode) {
  __shared__ __align__(16) half_t As[2][64 * LG];
  __shared__ __align__(16) half_t Bs[2][64 * LG];
  __shared__ __align__(16) float  Cs[64 * LS];
  __shared__ float redv[4][64];
  __shared__ float rscale[64];

  const int tid  = threadIdx.x;
  const int lane = tid & 31;
  const int wave = tid >> 5;
  const int wr = wave >> 1, wc = wave & 1;    // 4 x 2 wave grid
  const int m0 = blockIdx.y * 64;
  const int n0 = blockIdx.x * 64;
  const int sr = tid >> 2;                    // staging row 0..63
  const int sc = (tid & 3) * 8;               // staging col 0,8,16,24

  v8f c0 = {}, c1 = {};

#define GEMM_STAGE(kk_, b_)                                                   \
  do {                                                                        \
    async_b128(&A [(size_t)(m0 + sr) * CD + (kk_) + sc], &As[b_][sr * LG + sc]); \
    async_b128(&Wf[(size_t)(n0 + sr) * CD + (kk_) + sc], &Bs[b_][sr * LG + sc]); \
  } while (0)

  GEMM_STAGE(0, 0);
  int buf = 0;
  for (int kk = 0; kk < CD; kk += 32, buf ^= 1) {
    ASYNC_WAIT();
    __syncthreads();
    if (kk + 32 < CD) GEMM_STAGE(kk + 32, buf ^ 1);
    if (kk + 64 < CD) {
      __builtin_prefetch(&A [(size_t)(m0 + sr) * CD + kk + 64 + sc], 0, 1);
      __builtin_prefetch(&Wf[(size_t)(n0 + sr) * CD + kk + 64 + sc], 0, 1);
    }
    v16h a  = load_frag_a(&As[buf][wr * 16 * LG], LG, lane);
    v16h b0 = load_frag_b(&Bs[buf][(wc * 32 + 0)  * LG], LG, lane);
    v16h b1 = load_frag_b(&Bs[buf][(wc * 32 + 16) * LG], LG, lane);
    c0 = wmma_f16(a, b0, c0);
    c1 = wmma_f16(a, b1, c1);
  }
#undef GEMM_STAGE

  // C tiles (+bias) into LDS for the epilogue
  {
    int rb  = wr * 16 + ((lane >> 4) << 3);
    int nc0 = wc * 32 + (lane & 15);
#pragma unroll
    for (int i = 0; i < 8; ++i) {
      Cs[(rb + i) * LS + nc0]      = c0[i] + bias[n0 + nc0];
      Cs[(rb + i) * LS + nc0 + 16] = c1[i] + bias[n0 + nc0 + 16];
    }
  }
  __syncthreads();

  if (mode <= 1) {                            // per-head L2 norm, 4 thr / row
    int r = tid & 63, q = tid >> 6;
    float ps = 0.f;
#pragma unroll
    for (int i = 0; i < 16; ++i) { float v = Cs[r * LS + q * 16 + i]; ps += v * v; }
    redv[q][r] = ps;
    __syncthreads();
    if (q == 0) {
      float s = redv[0][r] + redv[1][r] + redv[2][r] + redv[3][r];
      rscale[r] = 1.0f / fmaxf(sqrtf(s), 1e-8f);
    }
    __syncthreads();
  }

#pragma unroll
  for (int i = 0; i < 16; ++i) {
    int flat = tid + 256 * i;
    int r = flat >> 6, c = flat & 63;
    float v = Cs[r * LS + c];
    int gm = m0 + r, gn = n0 + c;
    int b = gm >> 11, t = gm & 2047;
    int h = gn >> 6,  d = gn & 63;
    if (mode <= 1) {
      v *= rscale[r];
      outH[((size_t)((b * CH + h) * CT + t)) * CDK + d] = (half_t)v;
    } else if (mode == 2) {
      outH[((size_t)((b * CH + h) * CDK + d)) * CT + t] = (half_t)v;
    } else {
      outF[(size_t)gm * CD + gn] = 0.5f * v;
    }
  }
}

// ---------------------------------------------------------------------------
// Flash-style causal attention per (64-query tile, b*h). 8 waves, 64x64 tiles.
// Q,K: [bh][T][64] f16 (L2-normed).  Vt: [bh][64][T] f16.
// AO out: [b][t][h*64+d] f16.
// ---------------------------------------------------------------------------
__global__ __launch_bounds__(256) void attn64_kernel(
    const half_t* __restrict__ Q, const half_t* __restrict__ K,
    const half_t* __restrict__ Vt, const unsigned char* __restrict__ valid,
    half_t* __restrict__ AO) {
  __shared__ __align__(16) half_t Qs[64 * LQ];
  __shared__ __align__(16) half_t Ks[64 * LQ];
  __shared__ __align__(16) half_t Vs[64 * LQ];   // [d][t_local]
  __shared__ __align__(16) half_t Ps[64 * LQ];
  __shared__ __align__(16) float  Sf[64 * LS];
  __shared__ float redv[4][64];
  __shared__ float mrow[64], lrow[64], arow[64];

  const int tid  = threadIdx.x;
  const int lane = tid & 31;
  const int wave = tid >> 5;
  const int wr = wave >> 1, wc = wave & 1;
  const int bh  = blockIdx.y;                 // 0..31
  const int b   = bh >> 4;
  const int qt0 = blockIdx.x * 64;

  const half_t* Qbh = Q  + (size_t)bh * CT * CDK;
  const half_t* Kbh = K  + (size_t)bh * CT * CDK;
  const half_t* Vbh = Vt + (size_t)bh * CDK * CT;

  // Stage Q tile (raw; 1/sqrt(DK) applied to S after the WMMA)
#pragma unroll
  for (int ch = 0; ch < 2; ++ch) {
    int flat = tid + 256 * ch;                // 512 chunks of 8 halfs
    int r = flat >> 3, c8 = (flat & 7) * 8;
    async_b128(&Qbh[(size_t)(qt0 + r) * CDK + c8], &Qs[r * LQ + c8]);
  }
  if (tid < 64) { mrow[tid] = -3.0e38f; lrow[tid] = 0.f; }
  ASYNC_WAIT();
  __syncthreads();

  v8f o0 = {}, o1 = {};

  const int ktiles = blockIdx.x + 1;          // causal: key tile <= query tile
  for (int kt = 0; kt < ktiles; ++kt) {
    const int kt0 = kt * 64;
    // Stage K tile ([tok][d]) and V^T tile ([d][tok])
#pragma unroll
    for (int ch = 0; ch < 2; ++ch) {
      int flat = tid + 256 * ch;
      int r = flat >> 3, c8 = (flat & 7) * 8;
      async_b128(&Kbh[(size_t)(kt0 + r) * CDK + c8], &Ks[r * LQ + c8]);
      async_b128(&Vbh[(size_t)r * CT + kt0 + c8],    &Vs[r * LQ + c8]);
      if (kt + 1 < ktiles) {
        __builtin_prefetch(&Kbh[(size_t)(kt0 + 64 + r) * CDK + c8], 0, 1);
        __builtin_prefetch(&Vbh[(size_t)r * CT + kt0 + 64 + c8], 0, 1);
      }
    }
    ASYNC_WAIT();
    __syncthreads();

    // S = Q @ K^T  (reduction over DK = 64 -> two WMMA k-steps)
    v8f s0 = {}, s1 = {};
#pragma unroll
    for (int ks = 0; ks < 64; ks += 32) {
      v16h a  = load_frag_a(&Qs[wr * 16 * LQ + ks], LQ, lane);
      v16h b0 = load_frag_b(&Ks[(wc * 32 + 0)  * LQ + ks], LQ, lane);
      v16h b1 = load_frag_b(&Ks[(wc * 32 + 16) * LQ + ks], LQ, lane);
      s0 = wmma_f16(a, b0, s0);
      s1 = wmma_f16(a, b1, s1);
    }

    // scale -> clip(+-5) -> validity(-1e9) -> causal(+-1e9), spill fp32 LDS
    {
      int rb  = wr * 16 + ((lane >> 4) << 3);
      int nl0 = wc * 32 + (lane & 15);
#pragma unroll
      for (int hn = 0; hn < 2; ++hn) {
        int nl = nl0 + hn * 16;
        int keytok = kt0 + nl;
        bool ok = valid[b * CT + keytok] != 0;
#pragma unroll
        for (int i = 0; i < 8; ++i) {
          float s = ((hn == 0) ? s0[i] : s1[i]) * QK_SCALE;
          s = fminf(fmaxf(s, -5.f), 5.f);
          if (!ok) s = -1.0e9f;
          if (keytok > qt0 + rb + i) s += -1.0e9f;
          Sf[(rb + i) * LS + nl] = s;
        }
      }
    }
    __syncthreads();

    // Online softmax, 4 threads per query row (row = tid&63, 16-col chunks)
    {
      int r = tid & 63, q = tid >> 6;
      float pm = -3.4e38f;
#pragma unroll
      for (int i = 0; i < 16; ++i) pm = fmaxf(pm, Sf[r * LS + q * 16 + i]);
      redv[q][r] = pm;
      __syncthreads();
      float mo = mrow[r];
      float mx = fmaxf(fmaxf(fmaxf(redv[0][r], redv[1][r]),
                             fmaxf(redv[2][r], redv[3][r])), mo);
      __syncthreads();                        // WAR on redv
      float ps = 0.f;
#pragma unroll
      for (int i = 0; i < 16; ++i) {
        float p = __expf(Sf[r * LS + q * 16 + i] - mx);
        Ps[r * LQ + q * 16 + i] = (half_t)p;
        ps += p;
      }
      redv[q][r] = ps;
      __syncthreads();
      if (q == 0) {
        float sum = redv[0][r] + redv[1][r] + redv[2][r] + redv[3][r];
        float al = __expf(mo - mx);
        mrow[r] = mx;
        lrow[r] = lrow[r] * al + sum;
        arow[r] = al;
      }
      __syncthreads();
    }

    // Rescale accumulator, O += P @ V
    {
      int rb = wr * 16 + ((lane >> 4) << 3);
#pragma unroll
      for (int i = 0; i < 8; ++i) {
        float al = arow[rb + i];
        o0[i] *= al; o1[i] *= al;
      }
    }
#pragma unroll
    for (int ks = 0; ks < 64; ks += 32) {
      v16h a  = load_frag_a(&Ps[wr * 16 * LQ + ks], LQ, lane);
      v16h b0 = load_frag_b(&Vs[(wc * 32 + 0)  * LQ + ks], LQ, lane);
      v16h b1 = load_frag_b(&Vs[(wc * 32 + 16) * LQ + ks], LQ, lane);
      o0 = wmma_f16(a, b0, o0);
      o1 = wmma_f16(a, b1, o1);
    }
    __syncthreads();                          // Ks/Vs/Ps reused next tile
  }

  // O / l  -> attn-out f16 [b][t][h*64+d]
  {
    int rb  = wr * 16 + ((lane >> 4) << 3);
    int nl0 = wc * 32 + (lane & 15);
    int h = bh & 15;
#pragma unroll
    for (int i = 0; i < 8; ++i) {
      float inv = 1.0f / lrow[rb + i];
      int t = qt0 + rb + i;
      size_t base = ((size_t)(b * CT + t)) * CD + h * CDK;
      AO[base + nl0]      = (half_t)(o0[i] * inv);
      AO[base + nl0 + 16] = (half_t)(o1[i] * inv);
    }
  }
}

// ---------------------------------------------------------------------------
extern "C" void kernel_launch(void* const* d_in, const int* in_sizes, int n_in,
                              void* d_out, int out_size, void* d_ws, size_t ws_size,
                              hipStream_t stream) {
  const float* x  = (const float*)d_in[0];
  const float* Wq = (const float*)d_in[1];
  const float* bq = (const float*)d_in[2];
  const float* Wk = (const float*)d_in[3];
  const float* bk = (const float*)d_in[4];
  const float* Wv = (const float*)d_in[5];
  const float* bv = (const float*)d_in[6];
  const float* Wo = (const float*)d_in[7];
  const float* bo = (const float*)d_in[8];
  const unsigned char* vm = (const unsigned char*)d_in[9];  // bool mask [B,T]
  // d_in[10] = causal_mask: recomputed analytically in-kernel, never read.

  char* ws = (char*)d_ws;
  half_t* xh  = (half_t*)ws;  ws += (size_t)CM * CD * 2;            // 8 MB
  half_t* Wqh = (half_t*)ws;  ws += (size_t)CD * CD * 2;            // 2 MB
  half_t* Wkh = (half_t*)ws;  ws += (size_t)CD * CD * 2;
  half_t* Wvh = (half_t*)ws;  ws += (size_t)CD * CD * 2;
  half_t* Woh = (half_t*)ws;  ws += (size_t)CD * CD * 2;
  half_t* qh  = (half_t*)ws;  ws += (size_t)CM * CD * 2;            // 8 MB
  half_t* kh  = (half_t*)ws;  ws += (size_t)CM * CD * 2;
  half_t* vth = (half_t*)ws;  ws += (size_t)CM * CD * 2;
  half_t* aoh = (half_t*)ws;  ws += (size_t)CM * CD * 2;

  cvt_f32_to_f16_kernel<<<2048, 256, 0, stream>>>(x,  xh,  CM * CD);
  cvt_f32_to_f16_kernel<<<1024, 256, 0, stream>>>(Wq, Wqh, CD * CD);
  cvt_f32_to_f16_kernel<<<1024, 256, 0, stream>>>(Wk, Wkh, CD * CD);
  cvt_f32_to_f16_kernel<<<1024, 256, 0, stream>>>(Wv, Wvh, CD * CD);
  cvt_f32_to_f16_kernel<<<1024, 256, 0, stream>>>(Wo, Woh, CD * CD);

  dim3 gGemm(CD / 64, CM / 64);               // (16, 64)
  gemm64_kernel<<<gGemm, 256, 0, stream>>>(xh, Wqh, bq, qh,  nullptr, 0); // Q
  gemm64_kernel<<<gGemm, 256, 0, stream>>>(xh, Wkh, bk, kh,  nullptr, 1); // K
  gemm64_kernel<<<gGemm, 256, 0, stream>>>(xh, Wvh, bv, vth, nullptr, 2); // V^T

  dim3 gAttn(CT / 64, CB * CH);               // (32, 32)
  attn64_kernel<<<gAttn, 256, 0, stream>>>(qh, kh, vth, vm, aoh);

  gemm64_kernel<<<gGemm, 256, 0, stream>>>(aoh, Woh, bo, nullptr,
                                           (float*)d_out, 3);             // Wo
}
